// EViTEncoder_57440892617529
// MI455X (gfx1250) — matrix operand order repaired and checked by reading, not compiled, not run on hardware
//
#include <hip/hip_runtime.h>
#include <math.h>

// ---------------------------------------------------------------- types ----
typedef __bf16 bf16_t;
typedef __bf16 v16bf __attribute__((ext_vector_type(16)));
typedef float  v8f   __attribute__((ext_vector_type(8)));
typedef unsigned int v4u __attribute__((ext_vector_type(4)));
typedef int v4i __attribute__((ext_vector_type(4)));
typedef int v8i __attribute__((ext_vector_type(8)));

#if defined(__HIP_DEVICE_COMPILE__) && __has_builtin(__builtin_amdgcn_tensor_load_to_lds) && __has_builtin(__builtin_amdgcn_s_wait_tensorcnt)
#define HAVE_TDM 1
#else
#define HAVE_TDM 0
#endif

struct Geom {
  int Cin, Hin, Win, Ho, Wo, KH, KW, stride, pad, dil;
  int lgHW, lgWo, kk, invKK, invKW;  // power-of-two shifts + magic reciprocals
};

// ------------------------------------------------------- wmma fragments ----
// A fragment 16x32: lanes 0-15 row M=lane, K base 0; lanes 16-31 K base +16.
__device__ __forceinline__ v16bf ldAfrag(const bf16_t* s, int ld) {
  int lane = threadIdx.x & 31;
  int row  = lane & 15;
  int kb   = (lane >> 4) << 4;
  const bf16_t* r = s + row * ld;
  v16bf a;
#pragma unroll
  for (int j = 0; j < 16; ++j) a[j] = r[kb + j];
  return a;
}
__device__ __forceinline__ v16bf ldAfragF32(const float* s, int ld) {
  int lane = threadIdx.x & 31;
  int row  = lane & 15;
  int kb   = (lane >> 4) << 4;
  const float* r = s + row * ld;
  v16bf a;
#pragma unroll
  for (int j = 0; j < 16; ++j) a[j] = (bf16_t)r[kb + j];
  return a;
}
// B fragment 32x16 from K-major tile (B[k][n] = s[k*ld + n]).
__device__ __forceinline__ v16bf ldBfrag(const bf16_t* s, int ld) {
  int lane = threadIdx.x & 31;
  int col  = lane & 15;
  int kb   = (lane >> 4) << 4;
  v16bf b;
#pragma unroll
  for (int j = 0; j < 16; ++j) b[j] = s[(kb + j) * ld + col];
  return b;
}
// B fragment from N-major tile (B[k][n] = s[n*ld + k]) -> free transpose,
// per-lane contiguous reads. Used for TDM-staged weight rows and Q*K^T.
__device__ __forceinline__ v16bf ldBTfrag(const bf16_t* s, int ld) {
  int lane = threadIdx.x & 31;
  int col  = lane & 15;
  int kb   = (lane >> 4) << 4;
  const bf16_t* r = s + col * ld + kb;
  v16bf b;
#pragma unroll
  for (int j = 0; j < 16; ++j) b[j] = r[j];
  return b;
}
__device__ __forceinline__ v16bf ldBTfragF32(const float* s, int ld) {
  int lane = threadIdx.x & 31;
  int col  = lane & 15;
  int kb   = (lane >> 4) << 4;
  const float* r = s + col * ld + kb;
  v16bf b;
#pragma unroll
  for (int j = 0; j < 16; ++j) b[j] = (bf16_t)r[j];
  return b;
}

__device__ __forceinline__ v8f wmma_bf16(v16bf a, v16bf b, v8f c) {
  return __builtin_amdgcn_wmma_f32_16x16x32_bf16(false, a, false, b, (short)0,
                                                 c, false, false);
}
__device__ __forceinline__ float gelu_exact(float x) {
  return 0.5f * x * (1.0f + erff(x * 0.70710678118654752f));
}
__device__ __forceinline__ float epilogue(float v, int n, const float* bias,
                                          const float* scale,
                                          const float* shift, int act) {
  if (bias) v += bias[n];
  if (scale) v = v * scale[n] + shift[n];
  if (act == 1) v = gelu_exact(v);
  else if (act == 2) v = fmaxf(v, 0.f);
  return v;
}

// ----------------------------------------------------- TDM 2D tile load ----
// Row-gather: `rows` rows of `cols` fp32 elements, row stride `strideE`
// elements, zero-filled past (tenD0, tenD1). Lands contiguous in LDS.
#if HAVE_TDM
__device__ __forceinline__ void tdm_load_2d(const float* gsrc, unsigned ldsOff,
                                            int rows, int cols, int strideE,
                                            int tenD0, int tenD1) {
  unsigned long long ga = (unsigned long long)(uintptr_t)gsrc;
  v4u g0;
  g0[0] = 1u;                                   // count=1, user mode
  g0[1] = ldsOff;                               // lds_addr (bytes)
  g0[2] = (unsigned)(ga & 0xFFFFFFFFu);         // global_addr[31:0]
  g0[3] = (unsigned)((ga >> 32) & 0x1FFFFFFu) | (2u << 30);  // addr hi | type=2
  v8i g1;
  g1[0] = (2 << 16);                            // data_size = 4B
  g1[1] = (int)(((unsigned)tenD0 & 0xFFFFu) << 16);           // tensor_dim0 lo
  g1[2] = (int)(((unsigned)tenD0 >> 16) |
                (((unsigned)tenD1 & 0xFFFFu) << 16));         // d0 hi | d1 lo
  g1[3] = (int)((((unsigned)tenD1 >> 16) & 0xFFFFu) |
                ((unsigned)cols << 16));                      // d1 hi | tile_dim0
  g1[4] = rows;                                 // tile_dim1 (tile_dim2 = 0)
  g1[5] = strideE;                              // tensor_dim0_stride[31:0]
  g1[6] = 0;
  g1[7] = 0;
  v4i z4 = {0, 0, 0, 0};
#if defined(__clang_major__) && (__clang_major__ >= 23)
  v8i z8 = {0, 0, 0, 0, 0, 0, 0, 0};
  __builtin_amdgcn_tensor_load_to_lds(g0, g1, z4, z4, z8, 0);
#else
  __builtin_amdgcn_tensor_load_to_lds(g0, g1, z4, z4, 0);
#endif
}
#endif

// ----------------------------------------------- linear GEMM (hot path) ----
// out[m, noff+n] = epi( sum_k A[m,k] * Wt[n,k] ), A: [M x K] row-major.
// Block tile 128x64x32, wave tile 32x32 (4 wmma / K-step), TDM double-buffer.
#define LTM 128
#define LTN 64
#define LTK 32

__global__ void __launch_bounds__(256)
gemm_wmma(const float* __restrict__ A, const float* __restrict__ Wt,
          float* __restrict__ out, const float* __restrict__ bias,
          const float* __restrict__ scale, const float* __restrict__ shift,
          const float* __restrict__ addsrc, int M, int N, int K, int act,
          int ldo, int noff) {
  __shared__ float Asf[2][LTM][LTK];   // 32 KB  (row-major rows of A)
  __shared__ float Bsf[2][LTN][LTK];   // 16 KB  (weight rows, N-major)
  int m0 = blockIdx.x * LTM;
  int n0 = blockIdx.y * LTN;
  int tid = threadIdx.x, lane = tid & 31, w = tid >> 5;
  int wm = w & 3, wn = w >> 2;
  v8f acc00 = {}, acc01 = {}, acc10 = {}, acc11 = {};
  int nk = (K + LTK - 1) / LTK;
  int buf = 0;

#if HAVE_TDM
  if (w == 0) {
    tdm_load_2d(A + (size_t)m0 * K, (unsigned)(uintptr_t)&Asf[0][0][0], LTM,
                LTK, K, K, M - m0);
    tdm_load_2d(Wt + (size_t)n0 * K, (unsigned)(uintptr_t)&Bsf[0][0][0], LTN,
                LTK, K, K, N - n0);
  }
#endif
  for (int kt = 0; kt < nk; ++kt) {
    int k0 = kt * LTK;
#if HAVE_TDM
    if (w == 0) {
      if (kt + 1 < nk) {
        int k1 = k0 + LTK;
        tdm_load_2d(A + (size_t)m0 * K + k1,
                    (unsigned)(uintptr_t)&Asf[buf ^ 1][0][0], LTM, LTK, K,
                    K - k1, M - m0);
        tdm_load_2d(Wt + (size_t)n0 * K + k1,
                    (unsigned)(uintptr_t)&Bsf[buf ^ 1][0][0], LTN, LTK, K,
                    K - k1, N - n0);
        __builtin_amdgcn_s_wait_tensorcnt((short)2);
      } else {
        __builtin_amdgcn_s_wait_tensorcnt((short)0);
      }
    }
#else
    for (int i = tid; i < LTM * LTK; i += 256) {
      int r = i >> 5, c = i & 31;
      int m = m0 + r, k = k0 + c;
      Asf[buf][r][c] = (m < M && k < K) ? A[(size_t)m * K + k] : 0.f;
    }
    for (int i = tid; i < LTN * LTK; i += 256) {
      int r = i >> 5, c = i & 31;
      int n = n0 + r, k = k0 + c;
      Bsf[buf][r][c] = (n < N && k < K) ? Wt[(size_t)n * K + k] : 0.f;
    }
#endif
    __syncthreads();
    v16bf a0 = ldAfragF32(&Asf[buf][wm * 32][0], LTK);
    v16bf a1 = ldAfragF32(&Asf[buf][wm * 32 + 16][0], LTK);
    v16bf b0 = ldBTfragF32(&Bsf[buf][wn * 32][0], LTK);
    v16bf b1 = ldBTfragF32(&Bsf[buf][wn * 32 + 16][0], LTK);
    acc00 = wmma_bf16(a0, b0, acc00);
    acc01 = wmma_bf16(a0, b1, acc01);
    acc10 = wmma_bf16(a1, b0, acc10);
    acc11 = wmma_bf16(a1, b1, acc11);
    __syncthreads();
    buf ^= 1;
  }

  int rsel = (lane >= 16) ? 8 : 0;
  int nc = lane & 15;
#pragma unroll
  for (int mi = 0; mi < 2; ++mi) {
#pragma unroll
    for (int r = 0; r < 8; ++r) {
      int m = m0 + wm * 32 + mi * 16 + r + rsel;
      if (m >= M) continue;
#pragma unroll
      for (int ni = 0; ni < 2; ++ni) {
        int n = n0 + wn * 32 + ni * 16 + nc;
        if (n >= N) continue;
        v8f& acc = mi ? (ni ? acc11 : acc10) : (ni ? acc01 : acc00);
        float v = epilogue(acc[r], n, bias, scale, shift, act);
        if (addsrc) v += addsrc[(size_t)m * N + n];
        out[(size_t)m * ldo + noff + n] = v;
      }
    }
  }
}

// ------------------------------------------- im2col GEMM (patch embeds) ----
// amode 1 = gather from NCHW, amode 2 = gather from token-major NHWC.
#define GTM 64
#define GTN 64
#define GTK 32

__global__ void __launch_bounds__(256)
gemm_im2col(const float* __restrict__ A, const float* __restrict__ Wt,
            float* __restrict__ out, const float* __restrict__ bias, int M,
            int N, int K, int amode, Geom g) {
  __shared__ bf16_t As[GTM][GTK];
  __shared__ bf16_t Bs[GTK][GTN];
  int m0 = blockIdx.x * GTM;
  int n0 = blockIdx.y * GTN;
  int tid = threadIdx.x, lane = tid & 31, w = tid >> 5;
  int wm = w & 3, wn = w >> 2;
  v8f acc0 = {}, acc1 = {};
  int hwMask = (1 << g.lgHW) - 1, woMask = (1 << g.lgWo) - 1;

  for (int k0 = 0; k0 < K; k0 += GTK) {
    for (int i = tid; i < GTM * GTK; i += 256) {
      int r = i >> 5, c = i & 31;
      int m = m0 + r, k = k0 + c;
      float v = 0.f;
      if (m < M && k < K) {
        int b = m >> g.lgHW;
        int rr = m & hwMask;
        int oy = rr >> g.lgWo, ox = rr & woMask;
        int ci = (int)(((unsigned)k * (unsigned)g.invKK) >> 20);
        int rk = k - ci * g.kk;
        int ky = (int)(((unsigned)rk * (unsigned)g.invKW) >> 20);
        int kx = rk - ky * g.KW;
        int iy = oy * g.stride - g.pad + ky * g.dil;
        int ix = ox * g.stride - g.pad + kx * g.dil;
        if (iy >= 0 && iy < g.Hin && ix >= 0 && ix < g.Win) {
          size_t addr =
              (amode == 1)
                  ? (((size_t)(b * g.Cin + ci) * g.Hin + iy) * g.Win + ix)
                  : (((size_t)b * g.Hin * g.Win + (size_t)iy * g.Win + ix) *
                         g.Cin + ci);
          v = A[addr];
        }
      }
      As[r][c] = (bf16_t)v;
    }
    for (int i = tid; i < GTK * GTN; i += 256) {
      int kb = i >> 6, nb = i & 63;
      int n = n0 + nb, k = k0 + kb;
      float v = (n < N && k < K) ? Wt[(size_t)n * K + k] : 0.f;
      Bs[kb][nb] = (bf16_t)v;
    }
    __syncthreads();
    v16bf a  = ldAfrag(&As[wm * 16][0], GTK);
    v16bf b0 = ldBfrag(&Bs[0][wn * 32], GTN);
    v16bf b1 = ldBfrag(&Bs[0][wn * 32 + 16], GTN);
    acc0 = wmma_bf16(a, b0, acc0);
    acc1 = wmma_bf16(a, b1, acc1);
    __syncthreads();
  }
  int rsel = (lane >= 16) ? 8 : 0;
  int nc = lane & 15;
#pragma unroll
  for (int r = 0; r < 8; ++r) {
    int m = m0 + wm * 16 + r + rsel;
    if (m >= M) continue;
#pragma unroll
    for (int half = 0; half < 2; ++half) {
      int n = n0 + wn * 32 + half * 16 + nc;
      if (n >= N) continue;
      float v = (half ? acc1[r] : acc0[r]) + (bias ? bias[n] : 0.f);
      out[(size_t)m * N + n] = v;
    }
  }
}

// --------------------------------------------------------- attention -------
// One block: 64 queries of head (b,h). hd = 64, Mtok <= 256 KV tokens.
// Dynamic LDS: Qs 8KB | Kt 32KB (reused as P) | Vt 32KB | S 64KB = 136KB.
__global__ void __launch_bounds__(256)
attn_kernel(const float* __restrict__ q, const float* __restrict__ kv,
            float* __restrict__ o, int heads, int Ntok, int Mtok, int C,
            float sc) {
  extern __shared__ char smem[];
  bf16_t* Qs = (bf16_t*)smem;
  bf16_t* Kt = (bf16_t*)(smem + 8192);
  bf16_t* Vt = (bf16_t*)(smem + 8192 + 32768);
  float*  S  = (float*)(smem + 8192 + 65536);
  bf16_t* P  = Kt;
  const int hd = 64;
  int bh = blockIdx.y;
  int b = bh / heads, h = bh - b * heads;
  int n0 = blockIdx.x * 64;
  int tid = threadIdx.x, lane = tid & 31, w = tid >> 5;

  for (int i = tid; i < 64 * 64; i += 256) {
    int r = i >> 6, d = i & 63;
    int n = n0 + r;
    float v = (n < Ntok) ? q[((size_t)b * Ntok + n) * C + h * hd + d] : 0.f;
    Qs[r * 64 + d] = (bf16_t)v;
  }
  for (int i = tid; i < 256 * 64; i += 256) {
    int m = i >> 6, d = i & 63;
    float kvv = 0.f, vvv = 0.f;
    if (m < Mtok) {
      size_t base = ((size_t)b * Mtok + m) * (size_t)(2 * C);
      kvv = kv[base + h * hd + d];
      vvv = kv[base + C + h * hd + d];
    }
    Kt[m * 64 + d] = (bf16_t)kvv;
    Vt[m * 64 + d] = (bf16_t)vvv;
  }
  __syncthreads();

  int mt = w & 3;
  for (int nt = (w >> 2) * 8; nt < (w >> 2) * 8 + 8; ++nt) {
    v8f acc = {};
#pragma unroll
    for (int ks = 0; ks < 64; ks += 32) {
      v16bf a = ldAfrag(&Qs[(mt * 16) * 64 + ks], 64);
      v16bf bb = ldBTfrag(&Kt[(nt * 16) * 64 + ks], 64);
      acc = wmma_bf16(a, bb, acc);
    }
    int rsel = (lane >= 16) ? 8 : 0;
    int ncx = lane & 15;
#pragma unroll
    for (int r = 0; r < 8; ++r)
      S[(mt * 16 + r + rsel) * 256 + nt * 16 + ncx] = acc[r];
  }
  __syncthreads();

  if (tid < 64) {
    float mx = -3.4e38f;
    for (int c = 0; c < Mtok; ++c) mx = fmaxf(mx, S[tid * 256 + c]);
    float sum = 0.f;
    for (int c = 0; c < Mtok; ++c) {
      float e = __expf((S[tid * 256 + c] - mx) * sc);
      S[tid * 256 + c] = e;
      sum += e;
    }
    float inv = 1.f / sum;
    for (int c = 0; c < Mtok; ++c) S[tid * 256 + c] *= inv;
  }
  __syncthreads();

  for (int i = tid; i < 64 * 256; i += 256) {
    int cix = i & 255;
    P[i] = (bf16_t)((cix < Mtok) ? S[i] : 0.f);
  }
  __syncthreads();

#pragma unroll
  for (int t2 = 0; t2 < 2; ++t2) {
    int nt = (w >> 2) * 2 + t2;
    v8f acc = {};
#pragma unroll
    for (int ks = 0; ks < 256; ks += 32) {
      v16bf a = ldAfrag(&P[(mt * 16) * 256 + ks], 256);
      v16bf bb = ldBfrag(&Vt[ks * 64 + nt * 16], 64);
      acc = wmma_bf16(a, bb, acc);
    }
    int rsel = (lane >= 16) ? 8 : 0;
    int ncx = lane & 15;
#pragma unroll
    for (int r = 0; r < 8; ++r) {
      int n = n0 + mt * 16 + r + rsel;
      if (n < Ntok)
        o[((size_t)b * Ntok + n) * C + h * hd + nt * 16 + ncx] = acc[r];
    }
  }
}

// ------------------------------------------------------ small kernels ------
__global__ void ln_kernel(const float* __restrict__ in, float* __restrict__ out,
                          const float* __restrict__ g,
                          const float* __restrict__ b, int M, int C) {
  int m = blockIdx.x * blockDim.x + threadIdx.x;
  if (m >= M) return;
  const float* r = in + (size_t)m * C;
  float mean = 0.f;
  for (int c = 0; c < C; ++c) mean += r[c];
  mean /= C;
  float var = 0.f;
  for (int c = 0; c < C; ++c) {
    float d = r[c] - mean;
    var += d * d;
  }
  var /= C;
  float inv = rsqrtf(var + 1e-5f);
  float* o = out + (size_t)m * C;
  for (int c = 0; c < C; ++c) o[c] = (r[c] - mean) * inv * g[c] + b[c];
}

__global__ void dwconv_tok(const float* __restrict__ in,
                           const float* __restrict__ wt,
                           const float* __restrict__ bias,
                           float* __restrict__ out, int B, int C, int Hin,
                           int Win, int Ho, int Wo, int KH, int KW, int stride,
                           int pad, int dil) {
  size_t idx = (size_t)blockIdx.x * blockDim.x + threadIdx.x;
  size_t total = (size_t)B * Ho * Wo * C;
  if (idx >= total) return;
  int c = (int)(idx % C);
  size_t r = idx / C;
  int ox = (int)(r % Wo); r /= Wo;
  int oy = (int)(r % Ho);
  int b = (int)(r / Ho);
  float acc = bias ? bias[c] : 0.f;
  for (int ky = 0; ky < KH; ++ky) {
    int iy = oy * stride - pad + ky * dil;
    if (iy < 0 || iy >= Hin) continue;
    for (int kx = 0; kx < KW; ++kx) {
      int ix = ox * stride - pad + kx * dil;
      if (ix < 0 || ix >= Win) continue;
      acc += in[((size_t)b * Hin * Win + (size_t)iy * Win + ix) * C + c] *
             wt[(c * KH + ky) * KW + kx];
    }
  }
  out[((size_t)b * Ho * Wo + (size_t)oy * Wo + ox) * C + c] = acc;
}

__global__ void tok2nchw(const float* __restrict__ t, float* __restrict__ out,
                         int B, int C, int HW) {
  size_t idx = (size_t)blockIdx.x * blockDim.x + threadIdx.x;
  size_t total = (size_t)B * C * HW;
  if (idx >= total) return;
  int hw = (int)(idx % HW);
  size_t r = idx / HW;
  int c = (int)(r % C);
  int b = (int)(r / C);
  out[idx] = t[((size_t)b * HW + hw) * C + c];
}

__global__ void pool_tok(const float* __restrict__ t, float* __restrict__ out,
                         int B, int N, int C) {
  int idx = blockIdx.x * blockDim.x + threadIdx.x;
  if (idx >= B * C) return;
  int c = idx % C, b = idx / C;
  float s = 0.f;
  for (int n = 0; n < N; ++n) s += t[((size_t)b * N + n) * C + c];
  out[b * C + c] = s / (float)N;
}

__global__ void bcast_cat(const float* __restrict__ b4, float* __restrict__ cat,
                          int B, int N, int C, int ldc, int off) {
  size_t idx = (size_t)blockIdx.x * blockDim.x + threadIdx.x;
  size_t total = (size_t)B * N * C;
  if (idx >= total) return;
  int c = (int)(idx % C);
  size_t r = idx / C;
  int n = (int)(r % N);
  int b = (int)(r / N);
  cat[((size_t)b * N + n) * ldc + off + c] = b4[b * C + c];
}

// ------------------------------------------------------------- host --------
static const int kEMBED[4] = {64, 128, 256, 512};
static const int kHEADS[4] = {1, 2, 4, 8};
static const int kDEPTH[4] = {3, 4, 6, 3};
static const int kRR[4]    = {8, 4, 2, 1};

struct BlockP {
  const float *kv_w, *proj_b, *proj_w, *q_w;
  const float *sr_b, *sr_w, *srn_b, *srn_g;
  const float *dw_b, *dw_w, *fc1_b, *fc1_w, *fc2_b, *fc2_w;
  const float *n1_b, *n1_g, *n2_b, *n2_g;
};
struct StageP {
  BlockP blk[6];
  const float *ln_b, *ln_g, *norm_b, *norm_g, *patch_b, *patch_w;
};

static inline void launch_lin(hipStream_t s, const float* A, const float* Wt,
                              float* out, const float* bias,
                              const float* scale, const float* shift,
                              const float* add, int M, int N, int K, int act,
                              int ldo, int noff) {
  dim3 grid((M + LTM - 1) / LTM, (N + LTN - 1) / LTN);
  gemm_wmma<<<grid, 256, 0, s>>>(A, Wt, out, bias, scale, shift, add, M, N, K,
                                 act, ldo, noff);
}

extern "C" void kernel_launch(void* const* d_in, const int* in_sizes, int n_in,
                              void* d_out, int out_size, void* d_ws,
                              size_t ws_size, hipStream_t stream) {
  (void)in_sizes; (void)n_in; (void)out_size; (void)ws_size;
  auto F = [&](int i) { return (const float*)d_in[i]; };
  const float* x = F(0);
  int ii = 1;

  // ---- ASPP params (tree-flatten: alphabetical leaf order) ----
  const float *a_b1_b = F(ii++), *a_b1_g = F(ii++), *a_b1_w = F(ii++);
  const float *a_b2_b = F(ii++), *a_b2_dw = F(ii++), *a_b2_g = F(ii++),
              *a_b2_pw = F(ii++);
  const float *a_b3_b = F(ii++), *a_b3_dw = F(ii++), *a_b3_g = F(ii++),
              *a_b3_pw = F(ii++);
  const float *a_b4_b = F(ii++), *a_b4_g = F(ii++), *a_b4_w = F(ii++);
  const float *a_fu_b = F(ii++), *a_fu_g = F(ii++), *a_fu_w = F(ii++);

  // ---- stage params ----
  StageP st[4];
  for (int s = 0; s < 4; ++s) {
    for (int bi = 0; bi < kDEPTH[s]; ++bi) {
      BlockP& bp = st[s].blk[bi];
      bp.kv_w = F(ii++); bp.proj_b = F(ii++); bp.proj_w = F(ii++);
      bp.q_w = F(ii++);
      if (kRR[s] > 1) {
        bp.sr_b = F(ii++); bp.sr_w = F(ii++);
        bp.srn_b = F(ii++); bp.srn_g = F(ii++);
      } else {
        bp.sr_b = bp.sr_w = bp.srn_b = bp.srn_g = nullptr;
      }
      bp.dw_b = F(ii++); bp.dw_w = F(ii++);
      bp.fc1_b = F(ii++); bp.fc1_w = F(ii++);
      bp.fc2_b = F(ii++); bp.fc2_w = F(ii++);
      bp.n1_b = F(ii++); bp.n1_g = F(ii++);
      bp.n2_b = F(ii++); bp.n2_g = F(ii++);
    }
    st[s].ln_b = F(ii++); st[s].ln_g = F(ii++);
    st[s].norm_b = F(ii++); st[s].norm_g = F(ii++);
    st[s].patch_b = F(ii++); st[s].patch_w = F(ii++);
  }

  // ---- workspace bump allocator ----
  float* wsf = (float*)d_ws;
  size_t woff = 0;
  auto alloc = [&](size_t elems) {
    float* r = wsf + woff;
    woff += (elems + 63) & ~(size_t)63;
    return r;
  };
  float* bT  = alloc(8388608);
  float* bT2 = alloc(8388608);
  float* bU  = alloc(8388608);
  float* bQ  = alloc(8388608);
  float* bO  = alloc(8388608);
  float* bH  = alloc(33554432);
  float* bH2 = alloc(33554432);
  float* bXr = alloc(1 << 20);
  float* bKv = alloc(4 << 20);
  float* bPool = alloc(4096);
  float* bB4 = alloc(4096);

  float* outp = (float*)d_out;
  size_t outoff = 0;

  int Hin = 512, Win = 512, Cin = 3;
  float* t = bT;
  float* t2 = bT2;

  for (int s = 0; s < 4; ++s) {
    const StageP& sp = st[s];
    int C = kEMBED[s], R = kRR[s], heads = kHEADS[s], Hd = C * 4;
    int kz = (s == 0) ? 7 : 3, strd = (s == 0) ? 4 : 2, pad = kz / 2;
    int Hc = Hin / strd, Wc = Win / strd;
    int Ntok = Hc * Wc;
    int M = 8 * Ntok;

    // ---- patch embed (implicit-im2col WMMA GEMM) -> token-major ----
    Geom gp;
    gp.Cin = Cin; gp.Hin = Hin; gp.Win = Win; gp.Ho = Hc; gp.Wo = Wc;
    gp.KH = kz; gp.KW = kz; gp.stride = strd; gp.pad = pad; gp.dil = 1;
    gp.lgHW = __builtin_ctz(Hc * Wc); gp.lgWo = __builtin_ctz(Wc);
    gp.kk = kz * kz;
    gp.invKK = (1 << 20) / gp.kk + 1;
    gp.invKW = (1 << 20) / kz + 1;
    {
      dim3 grid((M + GTM - 1) / GTM, (C + GTN - 1) / GTN);
      gemm_im2col<<<grid, 256, 0, stream>>>((s == 0) ? x : t, sp.patch_w, t2,
                                            sp.patch_b, M, C, Cin * kz * kz,
                                            (s == 0) ? 1 : 2, gp);
    }
    { float* tmp = t; t = t2; t2 = tmp; }
    ln_kernel<<<(M + 255) / 256, 256, 0, stream>>>(t, t, sp.ln_g, sp.ln_b, M, C);

    // ---- transformer blocks ----
    for (int bi = 0; bi < kDEPTH[s]; ++bi) {
      const BlockP& bp = sp.blk[bi];
      ln_kernel<<<(M + 255) / 256, 256, 0, stream>>>(t, bU, bp.n1_g, bp.n1_b, M, C);
      launch_lin(stream, bU, bp.q_w, bQ, nullptr, nullptr, nullptr, nullptr,
                 M, C, C, 0, C, 0);
      const float* xrp = bU;
      int Mr = Ntok;
      if (R > 1) {
        int Hr = Hc / R, Wr = Wc / R;
        Mr = Hr * Wr;
        size_t tot = (size_t)8 * Mr * C;
        dwconv_tok<<<(unsigned)((tot + 255) / 256), 256, 0, stream>>>(
            bU, bp.sr_w, bp.sr_b, bXr, 8, C, Hc, Wc, Hr, Wr, R, R, R, 0, 1);
        ln_kernel<<<(8 * Mr + 255) / 256, 256, 0, stream>>>(bXr, bXr, bp.srn_g,
                                                            bp.srn_b, 8 * Mr, C);
        xrp = bXr;
      }
      launch_lin(stream, xrp, bp.kv_w, bKv, nullptr, nullptr, nullptr,
                 nullptr, 8 * Mr, 2 * C, C, 0, 2 * C, 0);
      dim3 agrid((Ntok + 63) / 64, 8 * heads);
      attn_kernel<<<agrid, 256, 139264, stream>>>(bQ, bKv, bO, heads, Ntok, Mr,
                                                  C, 0.125f);
      launch_lin(stream, bO, bp.proj_w, t2, bp.proj_b, nullptr, nullptr, t,
                 M, C, C, 0, C, 0);
      { float* tmp = t; t = t2; t2 = tmp; }

      ln_kernel<<<(M + 255) / 256, 256, 0, stream>>>(t, bU, bp.n2_g, bp.n2_b, M, C);
      launch_lin(stream, bU, bp.fc1_w, bH, bp.fc1_b, nullptr, nullptr,
                 nullptr, M, Hd, C, 1 /*gelu*/, Hd, 0);
      size_t tot = (size_t)M * Hd;
      dwconv_tok<<<(unsigned)((tot + 255) / 256), 256, 0, stream>>>(
          bH, bp.dw_w, bp.dw_b, bH2, 8, Hd, Hc, Wc, Hc, Wc, 3, 3, 1, 1, 1);
      launch_lin(stream, bH2, bp.fc2_w, t2, bp.fc2_b, nullptr, nullptr, t,
                 M, C, Hd, 0, C, 0);
      { float* tmp = t; t = t2; t2 = tmp; }
    }
    ln_kernel<<<(M + 255) / 256, 256, 0, stream>>>(t, t, sp.norm_g, sp.norm_b, M, C);

    // ---- ASPP after stage 3 (i == 2) ----
    if (s == 2) {
      float* cat = bH;   // M x 1024
      float* tmp = bH2;
      launch_lin(stream, t, a_b1_w, cat, nullptr, a_b1_g, a_b1_b, nullptr,
                 M, 256, 256, 2, 1024, 0);
      size_t tot = (size_t)M * 256;
      dwconv_tok<<<(unsigned)((tot + 255) / 256), 256, 0, stream>>>(
          t, a_b2_dw, nullptr, tmp, 8, 256, Hc, Wc, Hc, Wc, 3, 3, 1, 2, 2);
      launch_lin(stream, tmp, a_b2_pw, cat, nullptr, a_b2_g, a_b2_b, nullptr,
                 M, 256, 256, 2, 1024, 256);
      dwconv_tok<<<(unsigned)((tot + 255) / 256), 256, 0, stream>>>(
          t, a_b3_dw, nullptr, tmp, 8, 256, Hc, Wc, Hc, Wc, 3, 3, 1, 4, 4);
      launch_lin(stream, tmp, a_b3_pw, cat, nullptr, a_b3_g, a_b3_b, nullptr,
                 M, 256, 256, 2, 1024, 512);
      pool_tok<<<(8 * 256 + 255) / 256, 256, 0, stream>>>(t, bPool, 8, Ntok, 256);
      launch_lin(stream, bPool, a_b4_w, bB4, nullptr, a_b4_g, a_b4_b,
                 nullptr, 8, 256, 256, 2, 256, 0);
      bcast_cat<<<(unsigned)((tot + 255) / 256), 256, 0, stream>>>(
          bB4, cat, 8, Ntok, 256, 1024, 768);
      launch_lin(stream, cat, a_fu_w, t2, nullptr, a_fu_g, a_fu_b, nullptr,
                 M, 256, 1024, 2, 256, 0);
      { float* tswap = t; t = t2; t2 = tswap; }
    }

    // ---- emit NCHW feature map ----
    size_t tot = (size_t)8 * C * Ntok;
    tok2nchw<<<(unsigned)((tot + 255) / 256), 256, 0, stream>>>(
        t, outp + outoff, 8, C, Ntok);
    outoff += tot;

    Hin = Hc; Win = Wc; Cin = C;
  }
}
